// ReadoutBlock_9216999817872
// MI455X (gfx1250) — matrix operand order repaired and checked by reading, not compiled, not run on hardware
//
#include <hip/hip_runtime.h>
#include <hip/hip_bf16.h>

// ---------------- constants (match reference) ----------------
#define NNODE 20000
#define NEI   16
#define HDIM  256
#define HID   512   // MULT * H
#define NGR   64
#define ROWS  32    // rows per wave-tile (B fragments shared across 2 row-subtiles)
#define WAVES 2     // waves per block (block = 64 threads)

typedef __attribute__((ext_vector_type(16))) _Float16 v16h;
typedef __attribute__((ext_vector_type(8)))  _Float16 v8h;
typedef __attribute__((ext_vector_type(8)))  float    v8f;
typedef __attribute__((ext_vector_type(4)))  float    v4f;

static __device__ inline v16h cat16(v8h lo, v8h hi) {
    return __builtin_shufflevector(lo, hi, 0,1,2,3,4,5,6,7,8,9,10,11,12,13,14,15);
}

// gelu(tanh approx) == u * sigmoid(2t), t = 0.79788456*(u + 0.044715 u^3)
// branch-free with hardware exp2/rcp (v_exp_f32 / v_rcp_f32)
static __device__ inline float gelu_fast(float u) {
    float u2 = u * u;
    float s  = u * (1.5957691216057308f + 0.07135481627f * u2);   // 2t
    float e  = __builtin_amdgcn_exp2f(-1.4426950408889634f * s);  // exp(-2t)
    return u * __builtin_amdgcn_rcpf(1.0f + e);                   // u*sigmoid(2t)
}

// ---------------- generic LN -> FFN(GELU) -> residual kernel ----------------
// One wave per 32-row tile.  x: (R,256) fp32, out: (R,256) fp32.
// w1t: f16, column-major (512 cols x 256 K, K contiguous per col)
// w2t: f16, column-major (256 cols x 512 K, K contiguous per col)
__global__ __launch_bounds__(32 * WAVES)
void ffn_ln_wmma_kernel(const float* __restrict__ x, float* __restrict__ out,
                        const float* __restrict__ g, const float* __restrict__ b,
                        const _Float16* __restrict__ w1t, const float* __restrict__ b1,
                        const _Float16* __restrict__ w2t, const float* __restrict__ b2,
                        int R)
{
    extern __shared__ _Float16 smem[];  // per wave: 32*256 (xn) + 32*512 (hidden)
    const int lane = threadIdx.x & 31;
    const int wave = threadIdx.x >> 5;
    const int tile = blockIdx.x * WAVES + wave;
    const int row0 = tile * ROWS;
    if (row0 >= R) return;

    _Float16* xn = smem + wave * (ROWS * HDIM + ROWS * HID);
    _Float16* hb = xn + ROWS * HDIM;

    // ---------- LayerNorm: 32 rows, whole wave per row ----------
    const int c0 = lane * 8;                      // 8 columns per lane
    float gv[8], bv[8];
    #pragma unroll
    for (int j = 0; j < 8; j++) { gv[j] = g[c0 + j]; bv[j] = b[c0 + j]; }

    for (int m = 0; m < ROWS; m++) {
        int row = row0 + m; if (row >= R) row = R - 1;   // clamp keeps EXEC full
        const float* xr = x + (size_t)row * HDIM + c0;
        float v[8];
        *(v4f*)&v[0] = *(const v4f*)xr;
        *(v4f*)&v[4] = *(const v4f*)(xr + 4);
        float s = 0.f, q = 0.f;
        #pragma unroll
        for (int j = 0; j < 8; j++) { s += v[j]; q += v[j] * v[j]; }
        #pragma unroll
        for (int off = 16; off >= 1; off >>= 1) {
            s += __shfl_xor(s, off, 32);
            q += __shfl_xor(q, off, 32);
        }
        float mean = s * (1.0f / HDIM);
        float var  = q * (1.0f / HDIM) - mean * mean;
        float rstd = rsqrtf(var + 1e-5f);
        _Float16 o[8];
        #pragma unroll
        for (int j = 0; j < 8; j++)
            o[j] = (_Float16)((v[j] - mean) * rstd * gv[j] + bv[j]);
        *(v8h*)(xn + m * HDIM + c0) = *(v8h*)o;   // 16B DS store
    }
    asm volatile("s_wait_dscnt 0" ::: "memory");  // cross-lane LDS hand-off (in-order wave)

    // ---------- WMMA fragment index helpers (ISA 16-bit layouts) ----------
    const int mrow = lane & 15;               // A-matrix row (within subtile)
    const int ah   = (lane < 16) ? 0 : 8;     // A K-offset (first half); +16 for second
    const int bh   = (lane < 16) ? 0 : 16;    // unused: B fragments use ah, same k order as A
    const int lcol = lane & 15;

    // ---------- GEMM1: hidden = GELU(xn @ W1 + b1) -> LDS f16 ----------
    // 2 output col-tiles x 2 row-subtiles: each B fragment feeds 2 WMMAs
    for (int ntg = 0; ntg < HID / 16; ntg += 2) {
        v8f acc[2][2];                         // [colTile][rowSubtile]
        #pragma unroll
        for (int t = 0; t < 2; t++)
            #pragma unroll
            for (int h = 0; h < 2; h++) acc[t][h] = (v8f){};
        const _Float16* bcol[2];
        #pragma unroll
        for (int t = 0; t < 2; t++)
            bcol[t] = w1t + (size_t)((ntg + t) * 16 + lcol) * HDIM;

        #pragma unroll
        for (int ks = 0; ks < HDIM / 32; ks++) {
            const int kb = ks * 32;
            const _Float16* ap0 = xn + mrow * HDIM + kb + ah;
            const _Float16* ap1 = ap0 + 16 * HDIM;
            v16h a0 = cat16(*(const v8h*)ap0, *(const v8h*)(ap0 + 16));
            v16h a1 = cat16(*(const v8h*)ap1, *(const v8h*)(ap1 + 16));
            #pragma unroll
            for (int t = 0; t < 2; t++) {
                const _Float16* bp = bcol[t] + kb + ah;
                v16h bf = cat16(*(const v8h*)bp, *(const v8h*)(bp + 16));
                acc[t][0] = __builtin_amdgcn_wmma_f32_16x16x32_f16(false, a0, false, bf,
                                                                   (short)0, acc[t][0], false, false);
                acc[t][1] = __builtin_amdgcn_wmma_f32_16x16x32_f16(false, a1, false, bf,
                                                                   (short)0, acc[t][1], false, false);
            }
        }
        #pragma unroll
        for (int t = 0; t < 2; t++) {
            const int ncol = (ntg + t) * 16 + lcol;
            const float bias = b1[ncol];
            #pragma unroll
            for (int h = 0; h < 2; h++) {
                #pragma unroll
                for (int r = 0; r < 8; r++) {
                    const int mm = h * 16 + r + ah;      // C layout: M = r + (lane<16?0:8)
                    hb[mm * HID + ncol] = (_Float16)gelu_fast(acc[t][h][r] + bias);
                }
            }
        }
    }
    asm volatile("s_wait_dscnt 0" ::: "memory");

    // ---------- GEMM2: out = x + hidden @ W2 + b2 ----------
    for (int ntg = 0; ntg < HDIM / 16; ntg += 2) {
        v8f acc[2][2];
        #pragma unroll
        for (int t = 0; t < 2; t++)
            #pragma unroll
            for (int h = 0; h < 2; h++) acc[t][h] = (v8f){};
        const _Float16* bcol[2];
        #pragma unroll
        for (int t = 0; t < 2; t++)
            bcol[t] = w2t + (size_t)((ntg + t) * 16 + lcol) * HID;

        #pragma unroll
        for (int ks = 0; ks < HID / 32; ks++) {
            const int kb = ks * 32;
            const _Float16* ap0 = hb + mrow * HID + kb + ah;
            const _Float16* ap1 = ap0 + 16 * HID;
            v16h a0 = cat16(*(const v8h*)ap0, *(const v8h*)(ap0 + 16));
            v16h a1 = cat16(*(const v8h*)ap1, *(const v8h*)(ap1 + 16));
            #pragma unroll
            for (int t = 0; t < 2; t++) {
                const _Float16* bp = bcol[t] + kb + ah;
                v16h bf = cat16(*(const v8h*)bp, *(const v8h*)(bp + 16));
                acc[t][0] = __builtin_amdgcn_wmma_f32_16x16x32_f16(false, a0, false, bf,
                                                                   (short)0, acc[t][0], false, false);
                acc[t][1] = __builtin_amdgcn_wmma_f32_16x16x32_f16(false, a1, false, bf,
                                                                   (short)0, acc[t][1], false, false);
            }
        }
        #pragma unroll
        for (int t = 0; t < 2; t++) {
            const int ncol = (ntg + t) * 16 + lcol;
            const float bias = b2[ncol];
            #pragma unroll
            for (int h = 0; h < 2; h++) {
                #pragma unroll
                for (int r = 0; r < 8; r++) {
                    const int row = row0 + h * 16 + r + ah;
                    if (row < R) {
                        const size_t idx = (size_t)row * HDIM + ncol;
                        out[idx] = x[idx] + acc[t][h][r] + bias;
                    }
                }
            }
        }
    }
}

// ---------------- weight convert + transpose: w (K x Ncol, row-major fp32)
// -> wt (Ncol x K, K contiguous, f16) ----------------
__global__ void convert_wt_kernel(const float* __restrict__ w, _Float16* __restrict__ wt,
                                  int K, int Ncol)
{
    int t = blockIdx.x * blockDim.x + threadIdx.x;
    if (t < K * Ncol) {
        int k = t / Ncol, n = t % Ncol;
        wt[(size_t)n * K + k] = (_Float16)w[t];
    }
}

// ---------------- segment sum (atomics) ----------------
__global__ void seg_accum_kernel(const float* __restrict__ nf, const int* __restrict__ batch,
                                 float* __restrict__ seg, float* __restrict__ cnt, int n)
{
    int t = blockIdx.x * blockDim.x + threadIdx.x;
    int node = t >> 5;
    int part = t & 31;
    if (node < n) {
        int gi = batch[node];
        int c0 = part * 8;
        #pragma unroll
        for (int j = 0; j < 8; j++)
            atomicAdd(&seg[gi * HDIM + c0 + j], nf[(size_t)node * HDIM + c0 + j]);
        if (part == 0) atomicAdd(&cnt[gi], 1.0f);
    }
}

__global__ void seg_mean_kernel(const float* __restrict__ seg, const float* __restrict__ cnt,
                                float* __restrict__ gmean)
{
    int t = blockIdx.x * blockDim.x + threadIdx.x;
    if (t < NGR * HDIM)
        gmean[t] = seg[t] / fmaxf(cnt[t >> 8], 1.0f);
}

// ---------------- launch ----------------
extern "C" void kernel_launch(void* const* d_in, const int* in_sizes, int n_in,
                              void* d_out, int out_size, void* d_ws, size_t ws_size,
                              hipStream_t stream)
{
    const float* node_features = (const float*)d_in[0];
    const float* edge_features = (const float*)d_in[1];
    const int*   node_batch    = (const int*)d_in[2];
    // d_in[3] = num_graphs (== NGR, compile-time)

    const float* p[18];
    for (int i = 0; i < 18; i++) p[i] = (const float*)d_in[4 + i];
    // per set (node=0, global=1, edge=2): [6*s+0]=norm_g [1]=norm_b [2]=w1 [3]=b1 [4]=w2 [5]=b2

    // workspace layout
    char* ws = (char*)d_ws;
    const size_t WSZ = (size_t)HDIM * HID * sizeof(_Float16);   // 256 KB per matrix
    _Float16* w1t[3]; _Float16* w2t[3];
    for (int s = 0; s < 3; s++) {
        w1t[s] = (_Float16*)(ws + (size_t)(2 * s)     * WSZ);
        w2t[s] = (_Float16*)(ws + (size_t)(2 * s + 1) * WSZ);
    }
    float* seg   = (float*)(ws + 6 * WSZ);
    float* cnt   = seg + NGR * HDIM;
    float* gmean = cnt + NGR;

    hipMemsetAsync(seg, 0, (NGR * HDIM + NGR) * sizeof(float), stream);

    // convert weights to f16 transposed
    const int CE = HDIM * HID;
    for (int s = 0; s < 3; s++) {
        convert_wt_kernel<<<(CE + 255) / 256, 256, 0, stream>>>(p[6 * s + 2], w1t[s], HDIM, HID);
        convert_wt_kernel<<<(CE + 255) / 256, 256, 0, stream>>>(p[6 * s + 4], w2t[s], HID, HDIM);
    }

    // global features: scatter-mean
    seg_accum_kernel<<<(NNODE * 32 + 255) / 256, 256, 0, stream>>>(node_features, node_batch,
                                                                   seg, cnt, NNODE);
    seg_mean_kernel<<<(NGR * HDIM + 255) / 256, 256, 0, stream>>>(seg, cnt, gmean);

    float* out = (float*)d_out;
    float* out_global = out;                              // (64, 256)
    float* out_node   = out + NGR * HDIM;                 // (20000, 256)
    float* out_edge   = out_node + (size_t)NNODE * HDIM;  // (20000, 16, 256)

    const size_t LDSB = (size_t)WAVES * (ROWS * HDIM + ROWS * HID) * sizeof(_Float16); // 96 KB

    // global readout (set 1)
    {
        int tiles = (NGR + ROWS - 1) / ROWS, blocks = (tiles + WAVES - 1) / WAVES;
        ffn_ln_wmma_kernel<<<blocks, 32 * WAVES, LDSB, stream>>>(
            gmean, out_global, p[6], p[7], w1t[1], p[9], w2t[1], p[11], NGR);
    }
    // node readout (set 0)
    {
        int tiles = (NNODE + ROWS - 1) / ROWS, blocks = (tiles + WAVES - 1) / WAVES;
        ffn_ln_wmma_kernel<<<blocks, 32 * WAVES, LDSB, stream>>>(
            node_features, out_node, p[0], p[1], w1t[0], p[3], w2t[0], p[5], NNODE);
    }
    // edge readout (set 2): (N*NEI, H) rows
    {
        int R = NNODE * NEI;
        int tiles = (R + ROWS - 1) / ROWS, blocks = (tiles + WAVES - 1) / WAVES;
        ffn_ln_wmma_kernel<<<blocks, 32 * WAVES, LDSB, stream>>>(
            edge_features, out_edge, p[12], p[13], w1t[2], p[15], w2t[2], p[17], R);
    }
}
